// GSCLedge_14748917694890
// MI455X (gfx1250) — matrix operand order repaired and checked by reading, not compiled, not run on hardware
//
#include <hip/hip_runtime.h>
#include <hip/hip_bf16.h>
#include <hip/hip_fp16.h>

// Problem constants (match reference)
#define NN   8192
#define EE   262144
#define IND  128
#define HIDD 128
#define OUTD 64
#define BPAD 136   // padded LDS row stride in halfs (272 B) -> staggered banks
// 1/TEMP = 2.0

typedef __attribute__((ext_vector_type(16))) _Float16 v16h;
typedef __attribute__((ext_vector_type(8)))  _Float16 v8h;
typedef __attribute__((ext_vector_type(4)))  _Float16 v4h;
typedef __attribute__((ext_vector_type(8)))  float    v8f;

// ---------------- small dense GEMM: C = act(A@B + bias) ----------------
__global__ void k_gemm(const float* __restrict__ A, const float* __restrict__ B,
                       const float* __restrict__ bias, float* __restrict__ C,
                       int M, int K, int Nn, int act) {
  int idx = blockIdx.x * blockDim.x + threadIdx.x;
  if (idx >= M * Nn) return;
  int i = idx / Nn, j = idx - i * Nn;
  float s = 0.f;
  const float* a = A + (size_t)i * K;
  for (int k = 0; k < K; ++k) s = fmaf(a[k], B[(size_t)k * Nn + j], s);
  if (bias) s += bias[j];
  if (act) s = s > 0.f ? s : (__expf(s) - 1.f);   // ELU(alpha=1)
  C[idx] = s;
}

// ---------------- GCN helpers ----------------
__global__ void k_fill(float* p, float v, int n) {
  int i = blockIdx.x * blockDim.x + threadIdx.x;
  if (i < n) p[i] = v;
}

__global__ void k_deg(const int* __restrict__ dst, float* __restrict__ deg) {
  int e = blockIdx.x * blockDim.x + threadIdx.x;
  if (e < EE) atomicAdd(&deg[dst[e]], 1.0f);
}

__global__ void k_dinv(const float* __restrict__ deg, float* __restrict__ dinv) {
  int i = blockIdx.x * blockDim.x + threadIdx.x;
  if (i < NN) {
    float d = deg[i];
    dinv[i] = d > 0.f ? rsqrtf(d) : 0.f;
  }
}

// h = b + dinv[i]^2 * xw  (self-loop term folded with bias)
__global__ void k_selfinit(const float* __restrict__ xw, const float* __restrict__ dinv,
                           const float* __restrict__ bias, float* __restrict__ h) {
  int idx = blockIdx.x * blockDim.x + threadIdx.x;
  if (idx >= NN * HIDD) return;
  int i = idx >> 7, j = idx & 127;
  float di = dinv[i];
  h[idx] = bias[j] + di * di * xw[idx];
}

// one block per edge, 128 threads = one feature each
__global__ void k_scatter(const int* __restrict__ edge, const float* __restrict__ dinv,
                          const float* __restrict__ xw, float* __restrict__ h) {
  int e = blockIdx.x;
  int j = threadIdx.x;
  int src = edge[e];
  int dst = edge[EE + e];
  float nrm = dinv[src] * dinv[dst];
  atomicAdd(&h[(size_t)dst * HIDD + j], xw[(size_t)src * HIDD + j] * nrm);
}

// ---------------- L2 normalize (in place) + f16 convert ----------------
__global__ void k_normalize(float* __restrict__ z, _Float16* __restrict__ zh) {
  int row  = blockIdx.x * 4 + (threadIdx.x >> 5);
  int lane = threadIdx.x & 31;
  float4 v = *(const float4*)(z + (size_t)row * HIDD + lane * 4);
  float s = v.x * v.x + v.y * v.y + v.z * v.z + v.w * v.w;
  s += __shfl_xor(s, 1, 32);  s += __shfl_xor(s, 2, 32);
  s += __shfl_xor(s, 4, 32);  s += __shfl_xor(s, 8, 32);
  s += __shfl_xor(s, 16, 32);
  float inv = 1.0f / fmaxf(sqrtf(s), 1e-12f);
  v.x *= inv; v.y *= inv; v.z *= inv; v.w *= inv;
  *(float4*)(z + (size_t)row * HIDD + lane * 4) = v;
  v4h hv = { (_Float16)v.x, (_Float16)v.y, (_Float16)v.z, (_Float16)v.w };
  *(v4h*)(zh + (size_t)row * HIDD + lane * 4) = hv;
}

// ---------------- fused similarity GEMM + exp + row/col sums ----------------
// C[i,j] = exp(2 * (Ah_i . Bh_j)) reduced on the fly; N x N never materialized.
// 256 threads = 8 waves; block tile 128(M) x 64(N).
// B tile (64 x 128 f16, shared by all 8 waves) is staged into LDS once per
// block via the gfx1250 async LDS-DMA path (global_load_async_to_lds_b128,
// ASYNCcnt), then WMMA fragments are fed from ds_load_b128.
__global__ __launch_bounds__(256) void k_sim(const _Float16* __restrict__ Ah,
                                             const _Float16* __restrict__ Bh,
                                             float* __restrict__ rowsum,
                                             float* __restrict__ colsum) {
  __shared__ _Float16 Bs[64 * BPAD];

  const int tid  = threadIdx.x;
  const int lane = tid & 31;
  const int wave = tid >> 5;
  const int hlf  = lane >> 4;      // half-wave select
  const int r    = lane & 15;
  const int m0 = blockIdx.y * 128 + wave * 16;
  const int n0 = blockIdx.x * 64;

  const _Float16* arow = Ah + (size_t)(m0 + r) * HIDD;
  __builtin_prefetch(arow, 0, 0);            // global_prefetch for the A stream

  // ---- async-stage B tile: 64 rows x 128 halfs = 1024 x b128 chunks ----
  // LDS address is derived from &Bs[...] (ptrtoint) so the shared array
  // escapes into the asm: the "memory" clobber then makes the ds_load reads
  // below real (flat-LDS aperture maps addr[31:0] to the LDS offset).
#pragma unroll
  for (int c = tid; c < 1024; c += 256) {
    int row = c >> 4;                         // 0..63
    int col = (c & 15) << 3;                  // 0..120 halfs, step 8 (16 B)
    unsigned long long ga =
        (unsigned long long)(uintptr_t)(Bh + (size_t)(n0 + row) * HIDD + col);
    unsigned int la = (unsigned int)(uintptr_t)(Bs + row * BPAD + col);
    asm volatile("global_load_async_to_lds_b128 %0, %1, off"
                 :: "v"(la), "v"(ga) : "memory");
  }
  asm volatile("s_wait_asynccnt 0x0" ::: "memory");
  __syncthreads();

  v8f acc[4] = {};
#pragma unroll
  for (int ks = 0; ks < 128; ks += 32) {
    // A fragment (16x32 f16): lane half0 K={ks..ks+15}, half1 K={ks+16..ks+31}
    int abase = ks + hlf * 16;
    v8h alo = *(const v8h*)(arow + abase);
    v8h ahi = *(const v8h*)(arow + abase + 8);
    v16h a = __builtin_shufflevector(alo, ahi, 0, 1, 2, 3, 4, 5, 6, 7,
                                     8, 9, 10, 11, 12, 13, 14, 15);
    // B fragments from LDS: lane = column n0+j*16+r, contiguous K=16 per half
    int bbase = ks + hlf * 16;
#pragma unroll
    for (int j = 0; j < 4; ++j) {
      const _Float16* brow = Bs + (j * 16 + r) * BPAD + bbase;
      v8h blo = *(const v8h*)(brow);
      v8h bhi = *(const v8h*)(brow + 8);
      v16h b = __builtin_shufflevector(blo, bhi, 0, 1, 2, 3, 4, 5, 6, 7,
                                       8, 9, 10, 11, 12, 13, 14, 15);
      acc[j] = __builtin_amdgcn_wmma_f32_16x16x32_f16(
          false, a, false, b, (short)0, acc[j], false, false);
    }
  }

  // exp(2*s) then fold the four N-subtiles into one per-row partial
  v8f rs = {};
#pragma unroll
  for (int j = 0; j < 4; ++j) {
    float csum = 0.f;
#pragma unroll
    for (int v = 0; v < 8; ++v) {
      float e = __expf(acc[j][v] * 2.0f);
      rs[v] += e;
      csum += e;
    }
    if (colsum) {  // column sums (row-sums of the transposed 'between' matrix)
      csum += __shfl_xor(csum, 16, 32);
      if (lane < 16) atomicAdd(&colsum[n0 + j * 16 + lane], csum);
    }
  }
  // row sums: VGPR v holds row m0 + hlf*8 + v, N across the 16-lane group
#pragma unroll
  for (int v = 0; v < 8; ++v) {
    float t = rs[v];
    t += __shfl_xor(t, 1, 32);
    t += __shfl_xor(t, 2, 32);
    t += __shfl_xor(t, 4, 32);
    t += __shfl_xor(t, 8, 32);
    if (r == 0) atomicAdd(&rowsum[m0 + hlf * 8 + v], t);
  }
}

// ---------------- final per-node loss + mean ----------------
__global__ void k_loss(const float* __restrict__ z1, const float* __restrict__ z2,
                       const float* __restrict__ rs1, const float* __restrict__ rs2,
                       const float* __restrict__ rsb, const float* __restrict__ csb,
                       float* __restrict__ out) {
  int row  = blockIdx.x * 8 + (threadIdx.x >> 5);
  int lane = threadIdx.x & 31;
  float4 a = *(const float4*)(z1 + (size_t)row * HIDD + lane * 4);
  float4 b = *(const float4*)(z2 + (size_t)row * HIDD + lane * 4);
  float d12 = a.x * b.x + a.y * b.y + a.z * b.z + a.w * b.w;
  float d11 = a.x * a.x + a.y * a.y + a.z * a.z + a.w * a.w;
  float d22 = b.x * b.x + b.y * b.y + b.z * b.z + b.w * b.w;
#pragma unroll
  for (int off = 1; off < 32; off <<= 1) {
    d12 += __shfl_xor(d12, off, 32);
    d11 += __shfl_xor(d11, off, 32);
    d22 += __shfl_xor(d22, off, 32);
  }
  if (lane == 0) {
    float diag_b = __expf(2.f * d12);
    float x1 = rs1[row] + rsb[row] - __expf(2.f * d11);
    float x2 = rs2[row] + csb[row] - __expf(2.f * d22);
    float l1 = -__logf(diag_b / x1);
    float l2 = -__logf(diag_b / x2);
    atomicAdd(out, 0.5f * (l1 + l2) * (1.0f / (float)NN));
  }
}

// ---------------- host orchestration ----------------
extern "C" void kernel_launch(void* const* d_in, const int* in_sizes, int n_in,
                              void* d_out, int out_size, void* d_ws, size_t ws_size,
                              hipStream_t stream) {
  const int*   edge1 = (const int*)d_in[0];
  const int*   edge2 = (const int*)d_in[1];
  const float* feat1 = (const float*)d_in[2];
  const float* feat2 = (const float*)d_in[3];
  const float* W_gcn = (const float*)d_in[4];
  const float* b_gcn = (const float*)d_in[5];
  const float* fc1_W = (const float*)d_in[6];
  const float* fc1_b = (const float*)d_in[7];
  const float* fc2_W = (const float*)d_in[8];
  const float* fc2_b = (const float*)d_in[9];
  float* out = (float*)d_out;

  char* w = (char*)d_ws;
  auto alloc = [&](size_t bytes) -> void* {
    void* p = (void*)w;
    w += (bytes + 255) & ~(size_t)255;
    return p;
  };
  float*    xw1  = (float*)alloc((size_t)NN * HIDD * 4);
  float*    xw2  = (float*)alloc((size_t)NN * HIDD * 4);
  float*    h1   = (float*)alloc((size_t)NN * HIDD * 4);
  float*    h2   = (float*)alloc((size_t)NN * HIDD * 4);
  float*    zmid = (float*)alloc((size_t)NN * OUTD * 4);
  float*    z1   = (float*)alloc((size_t)NN * HIDD * 4);
  float*    z2   = (float*)alloc((size_t)NN * HIDD * 4);
  _Float16* z1h  = (_Float16*)alloc((size_t)NN * HIDD * 2);
  _Float16* z2h  = (_Float16*)alloc((size_t)NN * HIDD * 2);
  float*    deg  = (float*)alloc((size_t)NN * 4);
  float*    dinv = (float*)alloc((size_t)NN * 4);
  float*    sums = (float*)alloc((size_t)4 * NN * 4);  // rs1|rs2|rsb|csb
  float* rs1 = sums, *rs2 = sums + NN, *rsb = sums + 2 * NN, *csb = sums + 3 * NN;

  hipMemsetAsync(sums, 0, (size_t)4 * NN * 4, stream);
  hipMemsetAsync(out, 0, sizeof(float), stream);

  const int T = 256;
  const int gNH = (NN * HIDD + T - 1) / T;

  // ---- GCN view 1 ----
  k_gemm<<<gNH, T, 0, stream>>>(feat1, W_gcn, nullptr, xw1, NN, IND, HIDD, 0);
  k_fill<<<(NN + T - 1) / T, T, 0, stream>>>(deg, 1.0f, NN);          // self loops
  k_deg<<<(EE + T - 1) / T, T, 0, stream>>>(edge1 + EE, deg);         // in-degree by dst
  k_dinv<<<(NN + T - 1) / T, T, 0, stream>>>(deg, dinv);
  k_selfinit<<<gNH, T, 0, stream>>>(xw1, dinv, b_gcn, h1);
  k_scatter<<<EE, HIDD, 0, stream>>>(edge1, dinv, xw1, h1);

  // ---- GCN view 2 ----
  k_gemm<<<gNH, T, 0, stream>>>(feat2, W_gcn, nullptr, xw2, NN, IND, HIDD, 0);
  k_fill<<<(NN + T - 1) / T, T, 0, stream>>>(deg, 1.0f, NN);
  k_deg<<<(EE + T - 1) / T, T, 0, stream>>>(edge2 + EE, deg);
  k_dinv<<<(NN + T - 1) / T, T, 0, stream>>>(deg, dinv);
  k_selfinit<<<gNH, T, 0, stream>>>(xw2, dinv, b_gcn, h2);
  k_scatter<<<EE, HIDD, 0, stream>>>(edge2, dinv, xw2, h2);

  // ---- MLP projection ----
  const int gNO = (NN * OUTD + T - 1) / T;
  k_gemm<<<gNO, T, 0, stream>>>(h1, fc1_W, fc1_b, zmid, NN, HIDD, OUTD, 1);
  k_gemm<<<gNH, T, 0, stream>>>(zmid, fc2_W, fc2_b, z1, NN, OUTD, HIDD, 0);
  k_gemm<<<gNO, T, 0, stream>>>(h2, fc1_W, fc1_b, zmid, NN, HIDD, OUTD, 1);
  k_gemm<<<gNH, T, 0, stream>>>(zmid, fc2_W, fc2_b, z2, NN, OUTD, HIDD, 0);

  // ---- normalize + f16 convert ----
  k_normalize<<<NN / 4, 128, 0, stream>>>(z1, z1h);
  k_normalize<<<NN / 4, 128, 0, stream>>>(z2, z2h);

  // ---- fused WMMA similarity passes (the 51.5 GFLOP core) ----
  dim3 simGrid(NN / 64, NN / 128);
  k_sim<<<simGrid, 256, 0, stream>>>(z1h, z1h, rs1, nullptr);  // refl(z1)
  k_sim<<<simGrid, 256, 0, stream>>>(z2h, z2h, rs2, nullptr);  // refl(z2)
  k_sim<<<simGrid, 256, 0, stream>>>(z1h, z2h, rsb, csb);      // between + colsums

  // ---- final loss ----
  k_loss<<<NN / 8, 256, 0, stream>>>(z1, z2, rs1, rs2, rsb, csb, out);
}
